// PAM_Module_80272938762529
// MI455X (gfx1250) — matrix operand order repaired and checked by reading, not compiled, not run on hardware
//
#include <hip/hip_runtime.h>
#include <cstdint>

#define B_   8
#define C_   512
#define CQ_  64
#define N_   4096
#define BM   64
#define BN   64
#define NJ   (N_/BN)

typedef __attribute__((ext_vector_type(16))) __bf16 v16bf;
typedef __attribute__((ext_vector_type(8)))  __bf16 v8bf;
typedef __attribute__((ext_vector_type(8)))  float  v8f;
typedef __attribute__((ext_vector_type(4)))  unsigned int u32x4;
typedef __attribute__((ext_vector_type(8)))  int i32x8;
typedef __attribute__((ext_vector_type(4)))  int i32x4;

union AFrag  { v16bf v; v8bf h[2]; };
union BStore { short4 s4; __bf16 b[4]; };

__device__ __forceinline__ v8f wmma_bf16(v16bf a, v16bf b, v8f c) {
  return __builtin_amdgcn_wmma_f32_16x16x32_bf16(false, a, false, b, (short)0, c, false, false);
}

#if defined(__gfx1250__) && __has_builtin(__builtin_amdgcn_tensor_load_to_lds) && __has_builtin(__builtin_amdgcn_s_wait_tensorcnt)
#define USE_TDM 1
#else
#define USE_TDM 0
#endif

#if USE_TDM
// TDM descriptor per CDNA5 ISA 8.3/8.4: 2-D tile (64 x 512 rows) out of a
// (tensor_dim0=4096, stride=4096, tensor_dim1=512) bf16 tensor, packed into LDS.
// Toolchain uses the 6-arg builtin: (g0, g1, g2, g3, g4, cpol).
__device__ __forceinline__ void tdm_load_vtile(const __bf16* gsrc, unsigned lds_byte_off) {
  unsigned long long ga = (unsigned long long)(size_t)gsrc;
  u32x4 g0;
  g0[0] = 1u;                                   // count=1, user descriptor
  g0[1] = lds_byte_off;                         // lds_addr
  g0[2] = (unsigned)ga;                         // global_addr[31:0]
  g0[3] = (unsigned)(ga >> 32) | (2u << 30);    // global_addr[56:32] | type=2
  i32x8 g1;
  g1[0] = 0x10000;                              // data_size=1 (2 bytes), no mask/pad/iter
  g1[1] = (int)(4096u << 16);                   // tensor_dim0 = 4096 (low 16 in [63:48])
  g1[2] = (int)(512u  << 16);                   // tensor_dim1 = 512  (low 16 in [95:80])
  g1[3] = (int)(64u   << 16);                   // tile_dim0 = 64 ([127:112])
  g1[4] = 512;                                  // tile_dim1 = 512 ([143:128])
  g1[5] = 4096;                                 // tensor_dim0_stride = 4096 ([207:160] low)
  g1[6] = 0;
  g1[7] = 0;
  i32x4 z4 = {0, 0, 0, 0};                      // group2/group3: NULL for 2-D tensor
  i32x8 z8 = {0, 0, 0, 0, 0, 0, 0, 0};
  __builtin_amdgcn_tensor_load_to_lds(g0, g1, z4, z4, z8, 0);
}
#endif

// ---------------------------------------------------------------------------
// Kernel 1: fp32 -> bf16 weight conversion
// ---------------------------------------------------------------------------
__global__ __launch_bounds__(256) void cvt_bf16_kernel(const float* __restrict__ in,
                                                       __bf16* __restrict__ out, int n) {
  int i = blockIdx.x * blockDim.x + threadIdx.x;
  if (i < n) out[i] = (__bf16)in[i];
}

// ---------------------------------------------------------------------------
// Kernel 2: transpose x (b,c,n) fp32 -> XT (b,n,c) bf16 via LDS tiles
// ---------------------------------------------------------------------------
__global__ __launch_bounds__(256) void xpose_kernel(const float* __restrict__ x,
                                                    __bf16* __restrict__ XT) {
  __shared__ __bf16 tile[64][65];
  int bi = blockIdx.z, c0 = blockIdx.y * 64, n0 = blockIdx.x * 64;
  int t = threadIdx.x;
  int tg  = t >> 4;          // 0..15
  int tn4 = (t & 15) * 4;    // 0,4,...,60
#pragma unroll
  for (int rr = 0; rr < 4; ++rr) {
    int c = c0 + rr * 16 + tg;
    float4 v = *(const float4*)&x[((size_t)bi * C_ + c) * N_ + n0 + tn4];
    tile[tn4 + 0][rr * 16 + tg] = (__bf16)v.x;
    tile[tn4 + 1][rr * 16 + tg] = (__bf16)v.y;
    tile[tn4 + 2][rr * 16 + tg] = (__bf16)v.z;
    tile[tn4 + 3][rr * 16 + tg] = (__bf16)v.w;
  }
  __syncthreads();
#pragma unroll
  for (int rr = 0; rr < 4; ++rr) {
    int nl = rr * 16 + tg;
    BStore o;
    o.b[0] = tile[nl][tn4 + 0]; o.b[1] = tile[nl][tn4 + 1];
    o.b[2] = tile[nl][tn4 + 2]; o.b[3] = tile[nl][tn4 + 3];
    *(short4*)&XT[((size_t)bi * N_ + n0 + nl) * C_ + c0 + tn4] = o.s4;
  }
}

// ---------------------------------------------------------------------------
// Kernel 3: QKV projections with WMMA bf16.
//   Qb,Kb: (b, n, 64) row-major bf16.  Vb: (b, c, n) bf16.
// Block = 256 threads (8 waves), one 64-query slab, 160 D-tiles round-robin.
// ---------------------------------------------------------------------------
__global__ __launch_bounds__(256) void qkv_proj_kernel(
    const __bf16* __restrict__ XT,
    const __bf16* __restrict__ Wqb, const float* __restrict__ bq,
    const __bf16* __restrict__ Wkb, const float* __restrict__ bk,
    const __bf16* __restrict__ Wvb, const float* __restrict__ bv,
    __bf16* __restrict__ Qb, __bf16* __restrict__ Kb, __bf16* __restrict__ Vb) {
  int bi = blockIdx.y;
  int n0 = blockIdx.x * BM;
  int tid = threadIdx.x;
  int w = tid >> 5, lane = tid & 31, lh = lane >> 4, lc = lane & 15;
  const __bf16* X = XT + ((size_t)bi * N_ + n0) * C_;

  for (int T = w; T < 160; T += 8) {
    int kind, it, ot;
    const __bf16* W; const float* bias;
    if (T < 16)      { kind = 0; it = T >> 2;        ot = T & 3;        W = Wqb; bias = bq; }
    else if (T < 32) { kind = 1; it = (T - 16) >> 2; ot = (T - 16) & 3; W = Wkb; bias = bk; }
    else             { kind = 2; it = (T - 32) & 3;  ot = (T - 32) >> 2; W = Wvb; bias = bv; }

    v8f acc = {};
    const __bf16* xr = X + (size_t)(it * 16 + lc) * C_;     // A row = query n
    const __bf16* wr = W + (size_t)(ot * 16 + lc) * C_;     // B col = out channel
#pragma unroll 4
    for (int k0 = 0; k0 < C_; k0 += 32) {
      AFrag a;
      a.h[0] = *(const v8bf*)(xr + k0 + lh * 16);           // K {0..7}/{16..23}
      a.h[1] = *(const v8bf*)(xr + k0 + 8 + lh * 16);       // K {8..15}/{24..31}
      v16bf bf = *(const v16bf*)(wr + k0 + lh * 16);        // K {0..15}/{16..31}
      acc = wmma_bf16(a.v, bf, acc);
    }
    float bsc = bias[ot * 16 + lc];
    if (kind < 2) {
      __bf16* Dst = (kind == 0 ? Qb : Kb) + (size_t)bi * N_ * CQ_;
      int col = ot * 16 + lc;
#pragma unroll
      for (int r = 0; r < 8; ++r) {
        int n = n0 + it * 16 + 8 * lh + r;
        Dst[(size_t)n * CQ_ + col] = (__bf16)(acc[r] + bsc);
      }
    } else {
      int c = ot * 16 + lc;
      __bf16* Dst = Vb + ((size_t)bi * C_ + c) * N_ + n0 + it * 16 + 8 * lh;
      v8bf o;
#pragma unroll
      for (int r = 0; r < 8; ++r) o[r] = (__bf16)(acc[r] + bsc);
      *(v8bf*)Dst = o;
    }
  }
}

// ---------------------------------------------------------------------------
// Kernel 4: fused flash attention + gamma*out + x epilogue.
// Block = 512 threads (16 waves), 64 queries x all 512 channels.
// Pass 1: online row stats (m, l) with per-quarter partials.
// Pass 2: P = exp(S-m)/l in LDS; O += P * V^T with V tiles TDM-double-buffered.
// ---------------------------------------------------------------------------
__global__ __launch_bounds__(512) void attn_kernel(
    const __bf16* __restrict__ Qb, const __bf16* __restrict__ Kb,
    const __bf16* __restrict__ Vb, const float* __restrict__ xin,
    const float* __restrict__ gamma, float* __restrict__ outp) {
  extern __shared__ char smem[];
  __bf16* vbuf = (__bf16*)smem;                                        // [2][C_*BN]
  __bf16* pbuf = (__bf16*)(smem + (size_t)2 * C_ * BN * 2);            // [BM*BN]
  float*  pm   = (float*)(smem + ((size_t)2 * C_ * BN + BM * BN) * 2); // [4][BM]
  float*  pl   = pm + 4 * BM;                                          // [4][BM]
  float*  fm   = pl + 4 * BM;                                          // [BM]
  float*  fli  = fm + BM;                                              // [BM]

  int bi = blockIdx.y;
  int i0 = blockIdx.x * BM;
  int tid = threadIdx.x;
  int w = tid >> 5, lane = tid & 31, lh = lane >> 4, lc = lane & 15;

  const __bf16* Qbase = Qb + (size_t)bi * N_ * CQ_;
  const __bf16* Kbase = Kb + (size_t)bi * N_ * CQ_;
  const __bf16* Vbase = Vb + (size_t)bi * C_ * N_;

#if USE_TDM
  unsigned vbuf_off = (unsigned)(unsigned long long)(size_t)(void*)vbuf;
  if (w == 0) tdm_load_vtile(Vbase, vbuf_off);   // prefetch V tile 0 under pass 1
#endif

  // ---------------- Pass 1: softmax row statistics ----------------
  {
    int si = w & 3, q = w >> 2;                  // row sub-tile, j-quarter
    AFrag qa[2];
    const __bf16* qr = Qbase + (size_t)(i0 + si * 16 + lc) * CQ_;
    qa[0].h[0] = *(const v8bf*)(qr + lh * 16);
    qa[0].h[1] = *(const v8bf*)(qr + 8 + lh * 16);
    qa[1].h[0] = *(const v8bf*)(qr + 32 + lh * 16);
    qa[1].h[1] = *(const v8bf*)(qr + 40 + lh * 16);

    float m_r[8], l_r[8];
#pragma unroll
    for (int r = 0; r < 8; ++r) { m_r[r] = -3.0e38f; l_r[r] = 0.f; }

    for (int jb = q; jb < NJ; jb += 4) {
      int j0 = jb * BN;
      v8f S[4];
#pragma unroll
      for (int jt = 0; jt < 4; ++jt) {
        v8f acc = {};
        const __bf16* kr = Kbase + (size_t)(j0 + jt * 16 + lc) * CQ_;
        acc = wmma_bf16(qa[0].v, *(const v16bf*)(kr + lh * 16), acc);
        acc = wmma_bf16(qa[1].v, *(const v16bf*)(kr + 32 + lh * 16), acc);
        S[jt] = acc;
      }
#pragma unroll
      for (int r = 0; r < 8; ++r) {
        float v = fmaxf(fmaxf(S[0][r], S[1][r]), fmaxf(S[2][r], S[3][r]));
        for (int mk = 8; mk >= 1; mk >>= 1) v = fmaxf(v, __shfl_xor(v, mk, 16));
        float mn = fmaxf(m_r[r], v);
        float e = __expf(S[0][r] - mn) + __expf(S[1][r] - mn) +
                  __expf(S[2][r] - mn) + __expf(S[3][r] - mn);
        for (int mk = 8; mk >= 1; mk >>= 1) e += __shfl_xor(e, mk, 16);
        l_r[r] = l_r[r] * __expf(m_r[r] - mn) + e;
        m_r[r] = mn;
      }
    }
    if (lc == 0) {                               // lanes 0 and 16 publish partials
#pragma unroll
      for (int r = 0; r < 8; ++r) {
        int row = si * 16 + 8 * lh + r;
        pm[q * BM + row] = m_r[r];
        pl[q * BM + row] = l_r[r];
      }
    }
  }
  __syncthreads();
  if (tid < BM) {                                // merge the 4 quarter partials
    float m = -3.0e38f;
#pragma unroll
    for (int q2 = 0; q2 < 4; ++q2) m = fmaxf(m, pm[q2 * BM + tid]);
    float l = 0.f;
#pragma unroll
    for (int q2 = 0; q2 < 4; ++q2) l += pl[q2 * BM + tid] * __expf(pm[q2 * BM + tid] - m);
    fm[tid]  = m;
    fli[tid] = (l > 0.f) ? (1.f / l) : 0.f;
  }
  __syncthreads();

  // ---------------- Pass 2: O = softmax(S) * V^T ----------------
  int si2 = w >> 2, jt2 = w & 3;                 // S-tile assignment; it = si2
  AFrag qa2[2];
  {
    const __bf16* qr = Qbase + (size_t)(i0 + si2 * 16 + lc) * CQ_;
    qa2[0].h[0] = *(const v8bf*)(qr + lh * 16);
    qa2[0].h[1] = *(const v8bf*)(qr + 8 + lh * 16);
    qa2[1].h[0] = *(const v8bf*)(qr + 32 + lh * 16);
    qa2[1].h[1] = *(const v8bf*)(qr + 40 + lh * 16);
  }
  float sm[8], sli[8];
#pragma unroll
  for (int r = 0; r < 8; ++r) {
    int row = si2 * 16 + 8 * lh + r;
    sm[r] = fm[row]; sli[r] = fli[row];
  }

#if !USE_TDM
  {                                              // fallback: cooperative preload tile 0
    for (int row = tid; row < C_; row += 512) {
      const uint4* s4 = (const uint4*)(Vbase + (size_t)row * N_);
      uint4* d4 = (uint4*)(vbuf + row * BN);
#pragma unroll
      for (int k2 = 0; k2 < 8; ++k2) d4[k2] = s4[k2];
    }
    __syncthreads();
  }
#endif

  v8f oacc[8];
#pragma unroll
  for (int t = 0; t < 8; ++t) oacc[t] = (v8f){};

  for (int jb = 0; jb < NJ; ++jb) {
    int j0 = jb * BN;
    // S tile (si2, jt2) from global K (L2-resident, contiguous fragments)
    v8f s = {};
    {
      const __bf16* kr = Kbase + (size_t)(j0 + jt2 * 16 + lc) * CQ_;
      s = wmma_bf16(qa2[0].v, *(const v16bf*)(kr + lh * 16), s);
      s = wmma_bf16(qa2[1].v, *(const v16bf*)(kr + 32 + lh * 16), s);
    }
    __syncthreads();                             // everyone done with prev vbuf/pbuf reads

#if USE_TDM
    if (w == 0 && jb + 1 < NJ)
      tdm_load_vtile(Vbase + (size_t)(jb + 1) * BN,
                     vbuf_off + (unsigned)(((jb + 1) & 1) * (C_ * BN * 2)));
#else
    if (jb + 1 < NJ) {
      __bf16* dst = vbuf + (size_t)((jb + 1) & 1) * (C_ * BN);
      const __bf16* src = Vbase + (size_t)(jb + 1) * BN;
      for (int row = tid; row < C_; row += 512) {
        const uint4* s4 = (const uint4*)(src + (size_t)row * N_);
        uint4* d4 = (uint4*)(dst + row * BN);
#pragma unroll
        for (int k2 = 0; k2 < 8; ++k2) d4[k2] = s4[k2];
      }
    }
#endif
    // P = exp(S - m) / l  -> bf16 tile in LDS
    {
      int col = jt2 * 16 + lc;
#pragma unroll
      for (int r = 0; r < 8; ++r) {
        float p = __expf(s[r] - sm[r]) * sli[r];
        pbuf[(si2 * 16 + 8 * lh + r) * BN + col] = (__bf16)p;
      }
    }
#if USE_TDM
    if (w == 0) {
      if (jb + 1 < NJ) __builtin_amdgcn_s_wait_tensorcnt(1);  // current tile done
      else             __builtin_amdgcn_s_wait_tensorcnt(0);
    }
#endif
    __syncthreads();                             // P + current V tile visible

    // O += P * V^T : wave covers (it=si2) x 8 channel tiles
    AFrag pa[2];
    {
      const __bf16* pr = &pbuf[(si2 * 16 + lc) * BN];
      pa[0].h[0] = *(const v8bf*)(pr + lh * 16);
      pa[0].h[1] = *(const v8bf*)(pr + 8 + lh * 16);
      pa[1].h[0] = *(const v8bf*)(pr + 32 + lh * 16);
      pa[1].h[1] = *(const v8bf*)(pr + 40 + lh * 16);
    }
    const __bf16* vt = vbuf + (size_t)(jb & 1) * (C_ * BN);
#pragma unroll
    for (int t = 0; t < 8; ++t) {
      const __bf16* vr = vt + (((w & 3) * 8 + t) * 16 + lc) * BN;
      oacc[t] = wmma_bf16(pa[0].v, *(const v16bf*)(vr + lh * 16), oacc[t]);
      oacc[t] = wmma_bf16(pa[1].v, *(const v16bf*)(vr + 32 + lh * 16), oacc[t]);
    }
  }

  // ---------------- Epilogue: out = gamma * O + x ----------------
  float g = gamma[0];
#pragma unroll
  for (int t = 0; t < 8; ++t) {
    int c = ((w & 3) * 8 + t) * 16 + lc;
    size_t base = ((size_t)bi * C_ + c) * N_ + i0 + si2 * 16 + 8 * lh;
    const float4* xr = (const float4*)(xin + base);
    float4 o0, o1;
    o0.x = g * oacc[t][0] + xr[0].x;  o0.y = g * oacc[t][1] + xr[0].y;
    o0.z = g * oacc[t][2] + xr[0].z;  o0.w = g * oacc[t][3] + xr[0].w;
    o1.x = g * oacc[t][4] + xr[1].x;  o1.y = g * oacc[t][5] + xr[1].y;
    o1.z = g * oacc[t][6] + xr[1].z;  o1.w = g * oacc[t][7] + xr[1].w;
    float4* orow = (float4*)(outp + base);
    orow[0] = o0; orow[1] = o1;
  }
}

// ---------------------------------------------------------------------------
// Host launcher
// ---------------------------------------------------------------------------
extern "C" void kernel_launch(void* const* d_in, const int* in_sizes, int n_in,
                              void* d_out, int out_size, void* d_ws, size_t ws_size,
                              hipStream_t stream) {
  const float* x     = (const float*)d_in[0];
  const float* Wq    = (const float*)d_in[1];
  const float* bq    = (const float*)d_in[2];
  const float* Wk    = (const float*)d_in[3];
  const float* bk    = (const float*)d_in[4];
  const float* Wv    = (const float*)d_in[5];
  const float* bv    = (const float*)d_in[6];
  const float* gamma = (const float*)d_in[7];
  float* out = (float*)d_out;
  char* ws = (char*)d_ws;

  const size_t XT_OFF = 0;                                   // 32 MB
  const size_t QB_OFF = XT_OFF + (size_t)B_ * N_ * C_ * 2;   // 4 MB
  const size_t KB_OFF = QB_OFF + (size_t)B_ * N_ * CQ_ * 2;  // 4 MB
  const size_t VB_OFF = KB_OFF + (size_t)B_ * N_ * CQ_ * 2;  // 32 MB
  const size_t WQ_OFF = VB_OFF + (size_t)B_ * C_ * N_ * 2;
  const size_t WK_OFF = WQ_OFF + (size_t)CQ_ * C_ * 2;
  const size_t WV_OFF = WK_OFF + (size_t)CQ_ * C_ * 2;

  __bf16* XT  = (__bf16*)(ws + XT_OFF);
  __bf16* Qb  = (__bf16*)(ws + QB_OFF);
  __bf16* Kb  = (__bf16*)(ws + KB_OFF);
  __bf16* Vb  = (__bf16*)(ws + VB_OFF);
  __bf16* Wqb = (__bf16*)(ws + WQ_OFF);
  __bf16* Wkb = (__bf16*)(ws + WK_OFF);
  __bf16* Wvb = (__bf16*)(ws + WV_OFF);

  cvt_bf16_kernel<<<(CQ_ * C_ + 255) / 256, 256, 0, stream>>>(Wq, Wqb, CQ_ * C_);
  cvt_bf16_kernel<<<(CQ_ * C_ + 255) / 256, 256, 0, stream>>>(Wk, Wkb, CQ_ * C_);
  cvt_bf16_kernel<<<(C_ * C_ + 255) / 256, 256, 0, stream>>>(Wv, Wvb, C_ * C_);

  xpose_kernel<<<dim3(N_ / 64, C_ / 64, B_), 256, 0, stream>>>(x, XT);

  qkv_proj_kernel<<<dim3(N_ / BM, B_), 256, 0, stream>>>(
      XT, Wqb, bq, Wkb, bk, Wvb, bv, Qb, Kb, Vb);

  const unsigned smem_bytes =
      (2u * C_ * BN + BM * BN) * 2u + (4u * BM * 2u + 2u * BM) * 4u;  // 141824 B
  attn_kernel<<<dim3(N_ / BM, B_), 512, smem_bytes, stream>>>(
      Qb, Kb, Vb, x, gamma, out);
}
